// FractalEngine_25907242729867
// MI455X (gfx1250) — matrix operand order repaired and checked by reading, not compiled, not run on hardware
//
#include <hip/hip_runtime.h>
#include <math.h>

typedef __attribute__((ext_vector_type(16))) __bf16 v16bf;
typedef __attribute__((ext_vector_type(8)))  float  v8f;
typedef __attribute__((ext_vector_type(4)))  int    v4i;

#define DIN   1024
#define DHID  2048
#define DOUT  1024
#define DMLP  128
#define NROWS 8192
#define G3    6144  /* 3*DHID */
#define MB    2     /* row-tiles (16 rows each) per GRU block */

/* ---- workspace byte offsets (all >= 1KB aligned) ---- */
#define OFF_W1A   ((size_t)0)                                /* B-frags, K=2048 N=128  */
#define OFF_W1G   (OFF_W1A  + (size_t)2048*128*2)
#define OFF_W2A   (OFF_W1G  + (size_t)2048*128*2)            /* B-frags, K=128  N=1024 */
#define OFF_W2G   (OFF_W2A  + (size_t)128*1024*2)
#define OFF_WIH   (OFF_W2G  + (size_t)128*1024*2)            /* B-frags, K=1024 N=6144 */
#define OFF_WHH   (OFF_WIH  + (size_t)1024*6144*2)           /* B-frags, K=2048 N=6144 */
#define OFF_AHID  (OFF_WHH  + (size_t)2048*6144*2)           /* A-frags of hiddens     */
#define OFF_AOUT  (OFF_AHID + (size_t)NROWS*DHID*2)          /* A-frags of out         */
#define OFF_OUTB  (OFF_AOUT + (size_t)NROWS*DOUT*2)          /* out, bf16 row-major    */
#define OFF_TEN   (OFF_OUTB + (size_t)NROWS*DOUT*2)
#define OFF_MEAN  (OFF_TEN  + (size_t)NROWS*4)
#define OFF_GMEAN (OFF_MEAN + (size_t)8*DHID*4)
#define OFF_XW1   (OFF_GMEAN + (size_t)DHID*4)               /* xw1a[128] | xw1g[128]  */
#define OFF_PART  (OFF_XW1  + (size_t)256*4)                 /* 16 x 1024 partials     */
#define OFF_SCAL  (OFF_PART + (size_t)16*DOUT*4)

static __device__ __forceinline__ float sigf(float x) { return 1.0f / (1.0f + __expf(-x)); }

/* ---- async global->LDS staging (gfx1250 ASYNCcnt path), with safe fallback ---- */
#if defined(__AMDGCN__) && __has_builtin(__builtin_amdgcn_global_load_async_to_lds_b128)
#define HAS_ASYNC_LDS 1
typedef __attribute__((address_space(1))) v4i* as1_v4i;
typedef __attribute__((address_space(3))) v4i* as3_v4i;
#else
#define HAS_ASYNC_LDS 0
#endif

static __device__ __forceinline__ void stage16(const void* gsrc, void* ldst) {
#if HAS_ASYNC_LDS
  __builtin_amdgcn_global_load_async_to_lds_b128(
      (as1_v4i)(v4i*)const_cast<void*>(gsrc),
      (as3_v4i)(v4i*)ldst, 0, 0);
#else
  *(v4i*)ldst = *(const v4i*)gsrc;
#endif
}
static __device__ __forceinline__ void stage_wait() {
#if HAS_ASYNC_LDS
  asm volatile("s_wait_asynccnt 0" ::: "memory");
#endif
}

/* ---------- pack f32 row-major [K x N] weight into WMMA B fragments (bf16) ---------- */
__global__ void pack_b_kernel(const float* __restrict__ W, __bf16* __restrict__ dst,
                              int K, int N) {
  int idx = blockIdx.x * blockDim.x + threadIdx.x;
  if (idx >= K * N) return;
  int e  = idx & 15;
  int L  = (idx >> 4) & 31;
  int f  = idx >> 9;
  int KT = K >> 5;
  int nT = f / KT, kt = f - nT * KT;
  int k  = kt * 32 + ((e >> 3) << 4) + ((L >> 4) << 3) + (e & 7);
  int n  = nT * 16 + (L & 15);
  dst[idx] = (__bf16)W[(size_t)k * N + n];
}

/* ---------- pack f32 row-major [R x C] matrix into WMMA A fragments (bf16) ---------- */
__global__ void pack_a_kernel(const float* __restrict__ M, __bf16* __restrict__ dst,
                              int R, int C) {
  int idx = blockIdx.x * blockDim.x + threadIdx.x;
  if (idx >= R * C) return;
  int e  = idx & 15;
  int L  = (idx >> 4) & 31;
  int f  = idx >> 9;
  int KT = C >> 5;
  int rT = f / KT, kt = f - rT * KT;
  int r  = rT * 16 + (L & 15);
  int k  = kt * 32 + ((e >> 3) << 4) + ((L >> 4) << 3) + (e & 7);
  dst[idx] = (__bf16)M[(size_t)r * C + k];
}

/* ---------- same, but bf16 source (for out) ---------- */
__global__ void pack_a_bf_kernel(const __bf16* __restrict__ M, __bf16* __restrict__ dst,
                                 int R, int C) {
  int idx = blockIdx.x * blockDim.x + threadIdx.x;
  if (idx >= R * C) return;
  int e  = idx & 15;
  int L  = (idx >> 4) & 31;
  int f  = idx >> 9;
  int KT = C >> 5;
  int rT = f / KT, kt = f - rT * KT;
  int r  = rT * 16 + (L & 15);
  int k  = kt * 32 + ((e >> 3) << 4) + ((L >> 4) << 3) + (e & 7);
  dst[idx] = M[(size_t)r * C + k];
}

/* ---------- xw1[c] = x @ W1[:DIN, c] for both engines (broadcast row, computed once) */
__global__ void xw1_kernel(const float* __restrict__ x, const float* __restrict__ w1a,
                           const float* __restrict__ w1g, float* __restrict__ xw1) {
  int c = threadIdx.x;                 /* 256 threads: 0..127 -> a, 128..255 -> g */
  const float* W = (c < 128) ? w1a : w1g;
  int cc = c & 127;
  float s = 0.0f;
  for (int k = 0; k < DIN; ++k) s += x[k] * W[(size_t)k * DMLP + cc];
  xw1[c] = s;
}

/* ---------- engines: h1 = relu(hid@W1[DIN:] + xw1 + b1); out = h1a@W2a - h1g@W2g + db2
   dynamic LDS: 64KB of hiddens A-fragments (shared by all 8 waves) ---------- */
__global__ __launch_bounds__(256)
void engine_kernel(const v16bf* __restrict__ Ahid,
                   const v16bf* __restrict__ Bw1a, const v16bf* __restrict__ Bw1g,
                   const v16bf* __restrict__ Bw2a, const v16bf* __restrict__ Bw2g,
                   const float* __restrict__ xw1,
                   const float* __restrict__ b1a, const float* __restrict__ b1g,
                   const float* __restrict__ b2a, const float* __restrict__ b2g,
                   __bf16* __restrict__ outb, float* __restrict__ tens) {
  extern __shared__ char dynsm[];
  v16bf* Alds = (v16bf*)dynsm;                   /* (DHID/32)*32 = 2048 frags = 64KB */
  __shared__ __bf16 h1a[16 * DMLP];
  __shared__ __bf16 h1g[16 * DMLP];
  __shared__ float  tacc[16];

  const int lane = threadIdx.x & 31;
  const int wave = threadIdx.x >> 5;
  const int row0 = blockIdx.x * 16;
  const int m    = lane & 15;
  const int nn   = lane & 15;
  const int kA   = (lane >> 4) * 8;
  const int mOff = (lane >> 4) * 8;

  if (threadIdx.x < 16) tacc[threadIdx.x] = 0.0f;

  /* stage this row-tile's A fragments into LDS (async) */
  {
    const char* gsrc = (const char*)(Ahid + (size_t)blockIdx.x * (DHID / 32) * 32);
    for (int i = threadIdx.x; i < (DHID / 32) * 32 * 2; i += 256)
      stage16(gsrc + (size_t)i * 16, dynsm + (size_t)i * 16);
    stage_wait();
  }
  __syncthreads();

  /* layer 1: [16 x 128], K = 2048, one 16-col tile per wave */
  {
    const int col0 = wave * 16;
    v8f acca = {}, accg = {};
    for (int kt = 0; kt < DHID / 32; ++kt) {
      v16bf A  = Alds[kt * 32 + lane];
      v16bf Ba = Bw1a[((size_t)wave * (DHID / 32) + kt) * 32 + lane];
      v16bf Bg = Bw1g[((size_t)wave * (DHID / 32) + kt) * 32 + lane];
      acca = __builtin_amdgcn_wmma_f32_16x16x32_bf16(false, A, false, Ba, (short)0, acca, false, false);
      accg = __builtin_amdgcn_wmma_f32_16x16x32_bf16(false, A, false, Bg, (short)0, accg, false, false);
    }
    float ba = xw1[col0 + nn]       + b1a[col0 + nn];
    float bg = xw1[128 + col0 + nn] + b1g[col0 + nn];
#pragma unroll
    for (int v = 0; v < 8; ++v) {
      int mm = v + mOff;
      float fa = acca[v] + ba; fa = fa > 0.0f ? fa : 0.0f;
      float fg = accg[v] + bg; fg = fg > 0.0f ? fg : 0.0f;
      h1a[mm * DMLP + col0 + nn] = (__bf16)fa;
      h1g[mm * DMLP + col0 + nn] = (__bf16)fg;
    }
  }
  __syncthreads();

  /* layer 2: out[16 x 1024], K = 128 (A built from LDS) */
  for (int tt = wave; tt < DOUT / 16; tt += 8) {
    const int col0 = tt * 16;
    v8f acca = {}, accg = {};
#pragma unroll
    for (int kt = 0; kt < DMLP / 32; ++kt) {
      v16bf Aa, Ag;
#pragma unroll
      for (int v = 0; v < 8; ++v) {
        int k0 = kt * 32 + ((v >> 2) << 4) + kA + ((v & 3) << 1);
        Aa[2 * v]     = h1a[m * DMLP + k0];
        Aa[2 * v + 1] = h1a[m * DMLP + k0 + 1];
        Ag[2 * v]     = h1g[m * DMLP + k0];
        Ag[2 * v + 1] = h1g[m * DMLP + k0 + 1];
      }
      v16bf Ba = Bw2a[((size_t)tt * (DMLP / 32) + kt) * 32 + lane];
      v16bf Bg = Bw2g[((size_t)tt * (DMLP / 32) + kt) * 32 + lane];
      acca = __builtin_amdgcn_wmma_f32_16x16x32_bf16(false, Aa, false, Ba, (short)0, acca, false, false);
      accg = __builtin_amdgcn_wmma_f32_16x16x32_bf16(false, Ag, false, Bg, (short)0, accg, false, false);
    }
    float bb = b2a[col0 + nn] - b2g[col0 + nn];
#pragma unroll
    for (int v = 0; v < 8; ++v) {
      int mm = v + mOff;
      float o = acca[v] - accg[v] + bb;
      outb[(size_t)(row0 + mm) * DOUT + col0 + nn] = (__bf16)o;
      atomicAdd(&tacc[mm], o * o);
    }
  }
  __syncthreads();
  if (threadIdx.x < 16) tens[row0 + threadIdx.x] = tacc[threadIdx.x] * (1.0f / DOUT);
}

/* ---------- GRU, M-blocked x2: 32 rows/block so each B fragment feeds 2 WMMAs.
   dynamic LDS: 64KB out-A-frags + 128KB hiddens-A-frags = 192KB ---------- */
__global__ __launch_bounds__(256)
void gru_kernel(const float* __restrict__ hid,
                const v16bf* __restrict__ Aout, const v16bf* __restrict__ Ahid,
                const v16bf* __restrict__ Bwih, const v16bf* __restrict__ Bwhh,
                const float* __restrict__ tens, const float* __restrict__ wih_f32,
                const float* __restrict__ b_ih, const float* __restrict__ b_hh,
                float* __restrict__ hout) {
  extern __shared__ char dynsm[];
  v16bf* Ao = (v16bf*)dynsm;                          /* MB*1024 frags = 64KB  */
  v16bf* Ah = (v16bf*)(dynsm + MB * 32768);           /* MB*2048 frags = 128KB */

  const int lane = threadIdx.x & 31;
  const int wave = threadIdx.x >> 5;
  const int row0 = blockIdx.x * (16 * MB);
  const int nn   = lane & 15;
  const int mOff = (lane >> 4) * 8;
  const float* wlast = wih_f32 + (size_t)1024 * G3;

  {
    const char* go = (const char*)(Aout + (size_t)blockIdx.x * MB * (DOUT / 32) * 32);
    const char* gh = (const char*)(Ahid + (size_t)blockIdx.x * MB * (DHID / 32) * 32);
    for (int i = threadIdx.x; i < MB * (DOUT / 32) * 32 * 2; i += 256)
      stage16(go + (size_t)i * 16, dynsm + (size_t)i * 16);
    for (int i = threadIdx.x; i < MB * (DHID / 32) * 32 * 2; i += 256)
      stage16(gh + (size_t)i * 16, dynsm + MB * 32768 + (size_t)i * 16);
    stage_wait();
  }
  __syncthreads();

  for (int jj = wave; jj < DHID / 16; jj += 8) {
    v8f ai[3][MB] = {{{}, {}}, {{}, {}}, {{}, {}}};
    v8f ah[3][MB] = {{{}, {}}, {{}, {}}, {{}, {}}};

    /* gi: K over out (1024); one B load feeds MB row-tiles */
    for (int kt = 0; kt < DOUT / 32; ++kt) {
      v16bf A0 = Ao[kt * 32 + lane];
      v16bf A1 = Ao[1024 + kt * 32 + lane];
#pragma unroll
      for (int q = 0; q < 3; ++q) {
        v16bf B = Bwih[((size_t)(q * 128 + jj) * (DOUT / 32) + kt) * 32 + lane];
        ai[q][0] = __builtin_amdgcn_wmma_f32_16x16x32_bf16(false, A0, false, B, (short)0, ai[q][0], false, false);
        ai[q][1] = __builtin_amdgcn_wmma_f32_16x16x32_bf16(false, A1, false, B, (short)0, ai[q][1], false, false);
      }
    }
    /* gh: K over hiddens (2048) */
    for (int kt = 0; kt < DHID / 32; ++kt) {
      v16bf A0 = Ah[kt * 32 + lane];
      v16bf A1 = Ah[2048 + kt * 32 + lane];
#pragma unroll
      for (int q = 0; q < 3; ++q) {
        v16bf B = Bwhh[((size_t)(q * 128 + jj) * (DHID / 32) + kt) * 32 + lane];
        ah[q][0] = __builtin_amdgcn_wmma_f32_16x16x32_bf16(false, A0, false, B, (short)0, ah[q][0], false, false);
        ah[q][1] = __builtin_amdgcn_wmma_f32_16x16x32_bf16(false, A1, false, B, (short)0, ah[q][1], false, false);
      }
    }

    int c = jj * 16 + nn;
    float wl_r = wlast[c], wl_z = wlast[DHID + c], wl_n = wlast[2 * DHID + c];
    float bi_r = b_ih[c],  bi_z = b_ih[DHID + c],  bi_n = b_ih[2 * DHID + c];
    float bh_r = b_hh[c],  bh_z = b_hh[DHID + c],  bh_n = b_hh[2 * DHID + c];
#pragma unroll
    for (int rb = 0; rb < MB; ++rb) {
#pragma unroll
      for (int v = 0; v < 8; ++v) {
        int row = row0 + rb * 16 + v + mOff;
        float t   = tens[row];
        float ir  = ai[0][rb][v] + t * wl_r + bi_r;
        float iz  = ai[1][rb][v] + t * wl_z + bi_z;
        float in_ = ai[2][rb][v] + t * wl_n + bi_n;
        float hr  = ah[0][rb][v] + bh_r;
        float hz  = ah[1][rb][v] + bh_z;
        float hn  = ah[2][rb][v] + bh_n;
        float r   = sigf(ir + hr);
        float z   = sigf(iz + hz);
        float nc  = tanhf(in_ + r * hn);
        float ho  = hid[(size_t)row * DHID + c];
        hout[(size_t)row * DHID + c] = (1.0f - z) * nc + z * ho;
      }
    }
  }
}

/* ---------- faction means ---------- */
__global__ void means_kernel(const float* __restrict__ h, float* __restrict__ means) {
  int idx = blockIdx.x * blockDim.x + threadIdx.x;  /* 8*2048 */
  int f = idx >> 11, c = idx & 2047;
  const float* base = h + (size_t)f * 1024 * DHID + c;
  float s = 0.0f;
  for (int r = 0; r < 1024; ++r) s += base[(size_t)r * DHID];
  means[idx] = s * (1.0f / 1024.0f);
}

__global__ void gmean_kernel(const float* __restrict__ means, float* __restrict__ gmean) {
  int c = blockIdx.x * blockDim.x + threadIdx.x;
  if (c < DHID) {
    float s = 0.0f;
    for (int f = 0; f < 8; ++f) s += means[f * DHID + c];
    gmean[c] = s * 0.125f;
  }
}

/* ---------- in-place sync (+ debate if step>5) ---------- */
__global__ void blend_kernel(float* __restrict__ h, const float* __restrict__ means,
                             const float* __restrict__ gmean, const int* __restrict__ step) {
  unsigned idx = blockIdx.x * blockDim.x + threadIdx.x;  /* 8192*2048 */
  int i = idx / DHID, c = idx % DHID;
  int f = i >> 10;
  float v = h[idx];
  float r = 0.85f * v + 0.15f * means[f * DHID + c];
  if (*step > 5 && (i & 1023) < 256) r = 0.85f * r + 0.15f * gmean[c];
  h[idx] = r;
}

/* ---------- tension reduction: max, sumexp, avg ---------- */
__global__ void tred_kernel(const float* __restrict__ tens, float* __restrict__ scal,
                            float* __restrict__ d_avg) {
  __shared__ float sm[256];
  int tid = threadIdx.x;
  float mx = -1e30f;
  for (int i = tid; i < NROWS; i += 256) mx = fmaxf(mx, tens[i]);
  sm[tid] = mx; __syncthreads();
  for (int s = 128; s > 0; s >>= 1) { if (tid < s) sm[tid] = fmaxf(sm[tid], sm[tid + s]); __syncthreads(); }
  float gmax = sm[0]; __syncthreads();
  float se = 0.0f, st = 0.0f;
  for (int i = tid; i < NROWS; i += 256) { float t = tens[i]; se += __expf(t - gmax); st += t; }
  sm[tid] = se; __syncthreads();
  for (int s = 128; s > 0; s >>= 1) { if (tid < s) sm[tid] += sm[tid + s]; __syncthreads(); }
  float gse = sm[0]; __syncthreads();
  sm[tid] = st; __syncthreads();
  for (int s = 128; s > 0; s >>= 1) { if (tid < s) sm[tid] += sm[tid + s]; __syncthreads(); }
  if (tid == 0) { scal[0] = gmax; scal[1] = gse; d_avg[0] = sm[0] * (1.0f / NROWS); }
}

/* ---------- combined = softmax(t) @ out : stage 1, partial sums over 512-row chunks */
__global__ __launch_bounds__(256)
void combine_part_kernel(const __bf16* __restrict__ outb, const float* __restrict__ tens,
                         const float* __restrict__ scal, float* __restrict__ part) {
  __shared__ float w[512];  /* 2 KB */
  int rc = blockIdx.x >> 2;            /* row chunk 0..15  */
  int cb = blockIdx.x & 3;             /* col block 0..3   */
  int r0 = rc * 512;
  float gmax = scal[0];
  for (int i = threadIdx.x; i < 512; i += 256) w[i] = __expf(tens[r0 + i] - gmax);
  __syncthreads();
  int c = cb * 256 + threadIdx.x;
  float acc = 0.0f;
  for (int i = 0; i < 512; ++i) acc += w[i] * (float)outb[(size_t)(r0 + i) * DOUT + c];
  part[rc * DOUT + c] = acc;
}

/* ---------- stage 2: deterministic reduction over 16 partials ---------- */
__global__ void combine_red_kernel(const float* __restrict__ part,
                                   const float* __restrict__ scal, float* __restrict__ comb) {
  int c = blockIdx.x * blockDim.x + threadIdx.x;
  if (c < DOUT) {
    float acc = 0.0f;
    for (int rc = 0; rc < 16; ++rc) acc += part[rc * DOUT + c];
    comb[c] = acc / scal[1];
  }
}

extern "C" void kernel_launch(void* const* d_in, const int* in_sizes, int n_in,
                              void* d_out, int out_size, void* d_ws, size_t ws_size,
                              hipStream_t stream) {
  (void)in_sizes; (void)n_in; (void)out_size; (void)ws_size;
  const float* x    = (const float*)d_in[0];
  const float* hid  = (const float*)d_in[1];
  const float* w1a  = (const float*)d_in[2];
  const float* b1a  = (const float*)d_in[3];
  const float* w2a  = (const float*)d_in[4];
  const float* b2a  = (const float*)d_in[5];
  const float* w1g  = (const float*)d_in[6];
  const float* b1g  = (const float*)d_in[7];
  const float* w2g  = (const float*)d_in[8];
  const float* b2g  = (const float*)d_in[9];
  const float* wih  = (const float*)d_in[10];
  const float* bih  = (const float*)d_in[11];
  const float* whh  = (const float*)d_in[12];
  const float* bhh  = (const float*)d_in[13];
  const int*   step = (const int*)d_in[14];

  char* ws = (char*)d_ws;
  __bf16* w1a_b = (__bf16*)(ws + OFF_W1A);
  __bf16* w1g_b = (__bf16*)(ws + OFF_W1G);
  __bf16* w2a_b = (__bf16*)(ws + OFF_W2A);
  __bf16* w2g_b = (__bf16*)(ws + OFF_W2G);
  __bf16* wih_b = (__bf16*)(ws + OFF_WIH);
  __bf16* whh_b = (__bf16*)(ws + OFF_WHH);
  __bf16* ahid  = (__bf16*)(ws + OFF_AHID);
  __bf16* aout  = (__bf16*)(ws + OFF_AOUT);
  __bf16* outb  = (__bf16*)(ws + OFF_OUTB);
  float*  tens  = (float*)(ws + OFF_TEN);
  float*  means = (float*)(ws + OFF_MEAN);
  float*  gmean = (float*)(ws + OFF_GMEAN);
  float*  xw1   = (float*)(ws + OFF_XW1);
  float*  part  = (float*)(ws + OFF_PART);
  float*  scal  = (float*)(ws + OFF_SCAL);

  float* comb = (float*)d_out;
  float* avg  = (float*)d_out + 1024;
  float* hnew = (float*)d_out + 1025;

  /* one-time packing (weights reused 8192x; resident in 192MB L2) */
  xw1_kernel<<<1, 256, 0, stream>>>(x, w1a, w1g, xw1);
  pack_b_kernel<<<(2048 * 128) / 256, 256, 0, stream>>>(w1a + (size_t)DIN * DMLP, w1a_b, 2048, 128);
  pack_b_kernel<<<(2048 * 128) / 256, 256, 0, stream>>>(w1g + (size_t)DIN * DMLP, w1g_b, 2048, 128);
  pack_b_kernel<<<(128 * 1024) / 256, 256, 0, stream>>>(w2a, w2a_b, 128, 1024);
  pack_b_kernel<<<(128 * 1024) / 256, 256, 0, stream>>>(w2g, w2g_b, 128, 1024);
  pack_b_kernel<<<(1024 * G3) / 256, 256, 0, stream>>>(wih, wih_b, 1024, G3);
  pack_b_kernel<<<(2048 * G3) / 256, 256, 0, stream>>>(whh, whh_b, 2048, G3);
  pack_a_kernel<<<(NROWS * DHID) / 256, 256, 0, stream>>>(hid, ahid, NROWS, DHID);

  engine_kernel<<<NROWS / 16, 256, 65536, stream>>>((const v16bf*)ahid,
      (const v16bf*)w1a_b, (const v16bf*)w1g_b, (const v16bf*)w2a_b, (const v16bf*)w2g_b,
      xw1, b1a, b1g, b2a, b2g, outb, tens);

  pack_a_bf_kernel<<<(NROWS * DOUT) / 256, 256, 0, stream>>>(outb, aout, NROWS, DOUT);

  gru_kernel<<<NROWS / (16 * MB), 256, MB * 98304, stream>>>(hid, (const v16bf*)aout,
      (const v16bf*)ahid, (const v16bf*)wih_b, (const v16bf*)whh_b, tens, wih, bih, bhh, hnew);

  means_kernel<<<(8 * DHID) / 256, 256, 0, stream>>>(hnew, means);
  gmean_kernel<<<8, 256, 0, stream>>>(means, gmean);
  blend_kernel<<<(NROWS * DHID) / 256, 256, 0, stream>>>(hnew, means, gmean, step);
  tred_kernel<<<1, 256, 0, stream>>>(tens, scal, avg);
  combine_part_kernel<<<64, 256, 0, stream>>>(outb, tens, scal, part);
  combine_red_kernel<<<4, 256, 0, stream>>>(part, scal, comb);
}
